// MyLSTM_86681029967891
// MI455X (gfx1250) — matrix operand order repaired and checked by reading, not compiled, not run on hardware
//
#include <hip/hip_runtime.h>
#include <hip/hip_bf16.h>

// ---------------------------------------------------------------------------
// 2-layer LSTM (B=256, T=512, F=256, H=1024) + final projection (C=128).
// bf16 WMMA (f32 accumulate) for all GEMMs; weights packed once per call into
// the native V_WMMA_F32_16X16X32_BF16 B-fragment layout so the hot loop is
// pure 128-bit loads + WMMA. One kernel per (layer, timestep); h double-
// buffered, c updated in place, gates fused with the cell nonlinearity.
// ---------------------------------------------------------------------------

typedef __attribute__((ext_vector_type(16))) __bf16 v16bf;
typedef __attribute__((ext_vector_type(8)))  __bf16 v8bf;
typedef __attribute__((ext_vector_type(8)))  float  v8f;

#define B_ 256
#define T_ 512
#define F_ 256
#define H_ 1024
#define C_ 128

// ---------------------------------------------------------------- helpers --

__device__ __forceinline__ float sigmoid_f(float x) {
    return 1.0f / (1.0f + __expf(-x));
}

__device__ __forceinline__ v8f wmma_bf16(v16bf a, v16bf b, v8f c) {
    // (neg_a, A, neg_b, B, c_mod, C, reuse_a, reuse_b)
    return __builtin_amdgcn_wmma_f32_16x16x32_bf16(
        false, a, false, b, (short)0, c, false, false);
}

// A fragment (16x32 bf16) from a row-major bf16 matrix.
// ISA layout: lanes 0-15 (M=lane) hold K = kb..kb+7 and kb+16..kb+23 with
// kb = 32*kt; lanes 16-31 hold the +8 shifted halves.
__device__ __forceinline__ v16bf load_a_bf16(const __bf16* __restrict__ base,
                                             int strideK, int m0, int k0, int lane) {
    const int m  = m0 + (lane & 15);
    const int kb = k0 + ((lane >> 4) << 3);
    const __bf16* p = base + (size_t)m * strideK + kb;
    union { v16bf v; v8bf h[2]; } u;
    u.h[0] = *(const v8bf*)(p);        // K = kb .. kb+7
    u.h[1] = *(const v8bf*)(p + 16);   // K = kb+16 .. kb+23
    return u.v;
}

// Same A fragment but sourced from fp32 (converted in-register to bf16).
__device__ __forceinline__ v16bf load_a_f32(const float* __restrict__ base,
                                            int strideK, int m0, int k0, int lane) {
    const int m  = m0 + (lane & 15);
    const int kb = k0 + ((lane >> 4) << 3);
    const float* p = base + (size_t)m * strideK + kb;
    float4 a0 = *(const float4*)(p);
    float4 a1 = *(const float4*)(p + 4);
    float4 b0 = *(const float4*)(p + 16);
    float4 b1 = *(const float4*)(p + 20);
    union { v16bf v; __bf16 e[16]; } u;
    u.e[0]  = (__bf16)a0.x; u.e[1]  = (__bf16)a0.y;
    u.e[2]  = (__bf16)a0.z; u.e[3]  = (__bf16)a0.w;
    u.e[4]  = (__bf16)a1.x; u.e[5]  = (__bf16)a1.y;
    u.e[6]  = (__bf16)a1.z; u.e[7]  = (__bf16)a1.w;
    u.e[8]  = (__bf16)b0.x; u.e[9]  = (__bf16)b0.y;
    u.e[10] = (__bf16)b0.z; u.e[11] = (__bf16)b0.w;
    u.e[12] = (__bf16)b1.x; u.e[13] = (__bf16)b1.y;
    u.e[14] = (__bf16)b1.z; u.e[15] = (__bf16)b1.w;
    return u.v;
}

// B fragment (32x16 bf16) from the packed weight buffer: fragment
// (ntileG, kt) is 512 contiguous bf16 (32 lanes x 32B), lane-ordered.
__device__ __forceinline__ v16bf load_b_packed(const __bf16* __restrict__ pk,
                                               int nK, int ntileG, int kt, int lane) {
    const __bf16* q = pk + (((size_t)ntileG * nK + kt) << 9) + (lane << 4);
    union { v16bf v; v8bf h[2]; } u;
    u.h[0] = *(const v8bf*)(q);
    u.h[1] = *(const v8bf*)(q + 8);
    return u.v;
}

// ---------------------------------------------------------------- kernels --

// Pack fp32 weight W[rows x K] (rows mult of 16, K mult of 32) into bf16
// WMMA B-fragments. For fragment (ntile, kt): lane l, element e corresponds
// to W[ntile*16 + (l&15)][kt*32 + (l>>4)*8 + (e&7) + (e>>3)*16].
__global__ void pack_weight_kernel(const float* __restrict__ W,
                                   __bf16* __restrict__ dst, int K, int total) {
    int tid = blockIdx.x * blockDim.x + threadIdx.x;
    if (tid >= total) return;
    int frag   = tid >> 9;
    int within = tid & 511;
    int lane   = within >> 4;
    int e      = within & 15;
    int nK     = K >> 5;
    int ntile  = frag / nK;
    int kt     = frag - ntile * nK;
    int n = (ntile << 4) + (lane & 15);
    int k = (kt << 5) + ((lane >> 4) << 3) + (e & 7) + ((e >> 3) << 4);
    dst[tid] = (__bf16)W[(size_t)n * K + k];
}

// Zero all recurrent state (c fp32, h bf16 ping-pong buffers).
__global__ void init_state_kernel(float* __restrict__ c1, float* __restrict__ c2,
                                  __bf16* __restrict__ h1a, __bf16* __restrict__ h1b,
                                  __bf16* __restrict__ h2a, __bf16* __restrict__ h2b) {
    int i = blockIdx.x * blockDim.x + threadIdx.x;   // exactly B_*H_ threads
    c1[i] = 0.0f; c2[i] = 0.0f;
    __bf16 z = (__bf16)0.0f;
    h1a[i] = z; h1b[i] = z; h2a[i] = z; h2b[i] = z;
}

// One LSTM cell step for one layer. 1024 waves: wave -> (mtile, ntile) tile
// of the [256 x 1024] hidden state; 4 gate accumulators kept in VGPRs.
// XF32: x comes from fp32 (layer 1 reads y directly). OUT_TANH: layer 2's
// extra tanh on the recurrent output.
template<bool XF32, bool OUT_TANH>
__global__ __launch_bounds__(128)
void lstm_step_kernel(const float* __restrict__ x_f32,
                      const __bf16* __restrict__ x_bf,
                      int strideX, int nKx,
                      const __bf16* __restrict__ h_prev,
                      const __bf16* __restrict__ pWih,
                      const __bf16* __restrict__ pWhh,
                      const float* __restrict__ bih,
                      const float* __restrict__ bhh,
                      float* __restrict__ c_state,
                      __bf16* __restrict__ h_next) {
    const int lane   = threadIdx.x & 31;
    const int waveId = (blockIdx.x << 2) + (threadIdx.x >> 5);
    const int ntile  = waveId & 63;     // 1024/16 hidden-unit tiles
    const int mtile  = waveId >> 6;     // 256/16 batch tiles
    const int m0     = mtile << 4;
    const int col    = (ntile << 4) + (lane & 15);

    // Gate accumulators (i,f,g,o), seeded with the biases.
    v8f acc[4];
    #pragma unroll
    for (int g = 0; g < 4; ++g) {
        float bv = bih[g * H_ + col] + bhh[g * H_ + col];
        #pragma unroll
        for (int r = 0; r < 8; ++r) acc[g][r] = bv;
    }

    // Input contribution: x @ W_ih^T
    for (int kt = 0; kt < nKx; ++kt) {
        v16bf a;
        if constexpr (XF32) a = load_a_f32(x_f32, strideX, m0, kt * 32, lane);
        else                a = load_a_bf16(x_bf, strideX, m0, kt * 32, lane);
        #pragma unroll
        for (int g = 0; g < 4; ++g) {
            v16bf b = load_b_packed(pWih, nKx, (g << 6) + ntile, kt, lane);
            acc[g] = wmma_bf16(a, b, acc[g]);
        }
    }
    // Recurrent contribution: h @ W_hh^T
    for (int kt = 0; kt < (H_ / 32); ++kt) {
        v16bf a = load_a_bf16(h_prev, H_, m0, kt * 32, lane);
        #pragma unroll
        for (int g = 0; g < 4; ++g) {
            v16bf b = load_b_packed(pWhh, H_ / 32, (g << 6) + ntile, kt, lane);
            acc[g] = wmma_bf16(a, b, acc[g]);
        }
    }

    // Fused cell update. C/D layout: VGPR r -> M = r (lanes 0-15) / r+8
    // (lanes 16-31), N = lane&15.  Each (m,n) owned by exactly one lane.
    #pragma unroll
    for (int r = 0; r < 8; ++r) {
        int m = m0 + r + ((lane >> 4) << 3);
        size_t idx = (size_t)m * H_ + col;
        float iv = sigmoid_f(acc[0][r]);
        float fv = sigmoid_f(acc[1][r]);
        float gv = tanhf(acc[2][r]);
        float ov = sigmoid_f(acc[3][r]);
        float cn = fv * c_state[idx] + iv * gv;
        c_state[idx] = cn;
        float hn = ov * tanhf(cn);
        if constexpr (OUT_TANH) hn = tanhf(hn);
        h_next[idx] = (__bf16)hn;
    }
}

// Final projection: relu(h2 @ W_out^T + b_out) -> [256 x 128] fp32.
__global__ __launch_bounds__(128)
void out_proj_kernel(const __bf16* __restrict__ h,
                     const __bf16* __restrict__ pW,
                     const float* __restrict__ bias,
                     float* __restrict__ out) {
    const int lane   = threadIdx.x & 31;
    const int waveId = (blockIdx.x << 2) + (threadIdx.x >> 5);
    const int ntile  = waveId & 7;      // 128/16
    const int mtile  = waveId >> 3;     // 256/16
    const int m0     = mtile << 4;
    const int col    = (ntile << 4) + (lane & 15);

    float bv = bias[col];
    v8f acc;
    #pragma unroll
    for (int r = 0; r < 8; ++r) acc[r] = bv;

    for (int kt = 0; kt < (H_ / 32); ++kt) {
        v16bf a = load_a_bf16(h, H_, m0, kt * 32, lane);
        v16bf b = load_b_packed(pW, H_ / 32, ntile, kt, lane);
        acc = wmma_bf16(a, b, acc);
    }

    #pragma unroll
    for (int r = 0; r < 8; ++r) {
        int m = m0 + r + ((lane >> 4) << 3);
        out[(size_t)m * C_ + col] = fmaxf(acc[r], 0.0f);
    }
}

// ------------------------------------------------------------------ host --

extern "C" void kernel_launch(void* const* d_in, const int* in_sizes, int n_in,
                              void* d_out, int out_size, void* d_ws, size_t ws_size,
                              hipStream_t stream) {
    const float* y    = (const float*)d_in[0];
    const float* Wih1 = (const float*)d_in[1];
    const float* Whh1 = (const float*)d_in[2];
    const float* bih1 = (const float*)d_in[3];
    const float* bhh1 = (const float*)d_in[4];
    const float* Wih2 = (const float*)d_in[5];
    const float* Whh2 = (const float*)d_in[6];
    const float* bih2 = (const float*)d_in[7];
    const float* bhh2 = (const float*)d_in[8];
    const float* Wout = (const float*)d_in[9];
    const float* bout = (const float*)d_in[10];

    char* ws = (char*)d_ws;
    size_t off = 0;
    auto carve = [&](size_t bytes) -> void* {
        void* p = ws + off;
        off += (bytes + 255) & ~(size_t)255;
        return p;
    };
    __bf16* pWih1 = (__bf16*)carve((size_t)4 * H_ * F_ * 2);   //  2 MB
    __bf16* pWhh1 = (__bf16*)carve((size_t)4 * H_ * H_ * 2);   //  8 MB
    __bf16* pWih2 = (__bf16*)carve((size_t)4 * H_ * H_ * 2);   //  8 MB
    __bf16* pWhh2 = (__bf16*)carve((size_t)4 * H_ * H_ * 2);   //  8 MB
    __bf16* pWout = (__bf16*)carve((size_t)C_ * H_ * 2);       // .25 MB
    __bf16* h1buf[2] = { (__bf16*)carve((size_t)B_ * H_ * 2),
                         (__bf16*)carve((size_t)B_ * H_ * 2) };
    __bf16* h2buf[2] = { (__bf16*)carve((size_t)B_ * H_ * 2),
                         (__bf16*)carve((size_t)B_ * H_ * 2) };
    float* c1 = (float*)carve((size_t)B_ * H_ * 4);
    float* c2 = (float*)carve((size_t)B_ * H_ * 4);
    (void)ws_size; (void)in_sizes; (void)n_in; (void)out_size;

    const int thr = 256;
    // Pack weights (bf16 fragments, L2-resident afterwards: ~26 MB total).
    pack_weight_kernel<<<(4 * H_ * F_ + thr - 1) / thr, thr, 0, stream>>>(
        Wih1, pWih1, F_, 4 * H_ * F_);
    pack_weight_kernel<<<(4 * H_ * H_ + thr - 1) / thr, thr, 0, stream>>>(
        Whh1, pWhh1, H_, 4 * H_ * H_);
    pack_weight_kernel<<<(4 * H_ * H_ + thr - 1) / thr, thr, 0, stream>>>(
        Wih2, pWih2, H_, 4 * H_ * H_);
    pack_weight_kernel<<<(4 * H_ * H_ + thr - 1) / thr, thr, 0, stream>>>(
        Whh2, pWhh2, H_, 4 * H_ * H_);
    pack_weight_kernel<<<(C_ * H_ + thr - 1) / thr, thr, 0, stream>>>(
        Wout, pWout, H_, C_ * H_);

    // Fresh state every call (deterministic under graph replay).
    init_state_kernel<<<(B_ * H_) / thr, thr, 0, stream>>>(
        c1, c2, h1buf[0], h1buf[1], h2buf[0], h2buf[1]);

    // Sequential recurrence: one kernel per (layer, timestep).
    for (int t = 0; t < T_; ++t) {
        const float* xt = y + (size_t)t * F_;      // y[b, t, f]: row stride T*F
        lstm_step_kernel<true, false><<<256, 128, 0, stream>>>(
            xt, nullptr, T_ * F_, F_ / 32,
            h1buf[t & 1], pWih1, pWhh1, bih1, bhh1,
            c1, h1buf[(t + 1) & 1]);
        lstm_step_kernel<false, true><<<256, 128, 0, stream>>>(
            nullptr, h1buf[(t + 1) & 1], H_, H_ / 32,
            h2buf[t & 1], pWih2, pWhh2, bih2, bhh2,
            c2, h2buf[(t + 1) & 1]);
    }

    out_proj_kernel<<<32, 128, 0, stream>>>(h2buf[T_ & 1], pWout, bout,
                                            (float*)d_out);
}
